// LPE_48301202211102
// MI455X (gfx1250) — matrix operand order repaired and checked by reading, not compiled, not run on hardware
//
#include <hip/hip_runtime.h>

// ---------------- problem constants ----------------
#define WAYN 20
#define SHOTN 5
#define NQRY 1000
#define TN 8
#define DIMN 640
#define SEMN 300
#define NCLSN 351
#define NPIX 121
#define PPAD 128
#define NWT (WAYN * TN)   // 160

constexpr int A_RS = DIMN + 8;    // 648 bf16 per row (pad to rotate LDS banks)
constexpr int G_RS = PPAD + 8;    // 136
constexpr size_t SM_ABF = (size_t)PPAD * A_RS * 2;   // 165888
constexpr size_t SM_GT  = (size_t)PPAD * G_RS * 2;   // 34816
constexpr size_t SM_AT  = (size_t)NWT  * G_RS * 2;   // 43520
constexpr size_t SM_F32 = (size_t)(PPAD + 3 * NWT + 16) * 4; // invn+numer+dsq+sim+red
constexpr size_t SMEM_TOTAL = SM_ABF + SM_GT + SM_AT + SM_F32;

typedef __attribute__((ext_vector_type(16))) __bf16 v16bf;
typedef __attribute__((ext_vector_type(8)))  __bf16 v8bf;
typedef __attribute__((ext_vector_type(8)))  float  v8f;

// ---------------- small helpers ----------------
__device__ inline float block_reduce_sum(float v, float* scratch) {
    for (int o = 16; o > 0; o >>= 1) v += __shfl_down(v, o, 32);
    int lane = threadIdx.x & 31, wid = threadIdx.x >> 5;
    if (lane == 0) scratch[wid] = v;
    __syncthreads();
    if (threadIdx.x == 0) {
        float s = 0; int nw = (blockDim.x + 31) >> 5;
        for (int i = 0; i < nw; ++i) s += scratch[i];
        scratch[0] = s;
    }
    __syncthreads();
    float out = scratch[0];
    __syncthreads();
    return out;
}

__device__ inline float block_reduce_max(float v, float* scratch) {
    for (int o = 16; o > 0; o >>= 1) v = fmaxf(v, __shfl_down(v, o, 32));
    int lane = threadIdx.x & 31, wid = threadIdx.x >> 5;
    if (lane == 0) scratch[wid] = v;
    __syncthreads();
    if (threadIdx.x == 0) {
        float s = -3.0e38f; int nw = (blockDim.x + 31) >> 5;
        for (int i = 0; i < nw; ++i) s = fmaxf(s, scratch[i]);
        scratch[0] = s;
    }
    __syncthreads();
    float out = scratch[0];
    __syncthreads();
    return out;
}

__device__ inline float sigmoidf(float x) { return 1.0f / (1.0f + expf(-x)); }
__device__ inline float lreluf(float x)   { return x >= 0.0f ? x : 0.1f * x; }

// ---------------- WMMA fragment loaders (bf16, 16x16x32) ----------------
// A fragment (16x32, MxK): lanes 0-15 row M=lane, K base 0; lanes 16-31 same rows, K base 8.
// Per lane: K=[kb,kb+8) -> regs 0..7 of v16bf low half, K=[kb+16,kb+24) -> high half.
__device__ inline v16bf frag_combine(v8bf lo, v8bf hi) {
    v16bf r;
#pragma unroll
    for (int i = 0; i < 8; ++i) { r[i] = lo[i]; r[i + 8] = hi[i]; }
    return r;
}

__device__ inline v16bf load_frag_a(const __bf16* base, int row0, int k0, int rs, int lane) {
    const __bf16* p = base + (size_t)(row0 + (lane & 15)) * rs + k0 + ((lane >> 4) << 3);
    v8bf lo = *(const v8bf*)p;
    v8bf hi = *(const v8bf*)(p + 16);
    return frag_combine(lo, hi);
}

// B fragment (32x16, KxN) from a [col][K] row-major array:
// lanes 0-15: col=lane, K base 0; lanes 16-31: col=lane-16, K base 8; K=[kb,kb+8) low half, [kb+16,kb+24) high half.
__device__ inline v16bf load_frag_b(const __bf16* base, int col0, int k0, int rs, int lane) {
    const __bf16* p = base + (size_t)(col0 + (lane & 15)) * rs + k0 + ((lane >> 4) << 3);
    v8bf lo = *(const v8bf*)p;
    v8bf hi = *(const v8bf*)(p + 16);
    return frag_combine(lo, hi);
}

// ---------------- stage kernels ----------------

// filters: h = lrelu(l2v@w1^T+b1); filt = h@w2^T+b2; L2n -> filtn (f32) + Wglob rows [w*T+t] (bf16)
__global__ void k_filt(const float* __restrict__ l2v, const float* __restrict__ w1,
                       const float* __restrict__ b1, const float* __restrict__ w2,
                       const float* __restrict__ b2, float* __restrict__ filtn,
                       __bf16* __restrict__ wglob) {
    int t = blockIdx.x % TN, w = blockIdx.x / TN;
    __shared__ float hloc[SEMN];
    __shared__ float floc[DIMN];
    __shared__ float red[16];
    for (int d = threadIdx.x; d < SEMN; d += blockDim.x) {
        float acc = b1[t * SEMN + d];
        const float* wr = w1 + ((size_t)t * SEMN + d) * SEMN;
        const float* lv = l2v + (size_t)w * SEMN;
        for (int s = 0; s < SEMN; ++s) acc += lv[s] * wr[s];
        hloc[d] = lreluf(acc);
    }
    __syncthreads();
    float ssl = 0;
    for (int c = threadIdx.x; c < DIMN; c += blockDim.x) {
        float acc = b2[t * DIMN + c];
        const float* wr = w2 + ((size_t)t * DIMN + c) * SEMN;
        for (int d = 0; d < SEMN; ++d) acc += hloc[d] * wr[d];
        floc[c] = acc;
        ssl += acc * acc;
    }
    float ss = block_reduce_sum(ssl, red);
    float inv = 1.0f / fmaxf(sqrtf(ss), 1e-12f);
    for (int c = threadIdx.x; c < DIMN; c += blockDim.x) {
        float v = floc[c] * inv;
        filtn[((size_t)t * WAYN + w) * DIMN + c] = v;
        wglob[((size_t)w * TN + t) * DIMN + c] = (__bf16)v;
    }
}

// template weights wgt[w][t]
__global__ void k_wgt(const float* __restrict__ l2v, const float* __restrict__ wt1,
                      const float* __restrict__ bt1, const float* __restrict__ wt2,
                      const float* __restrict__ bt2, const float* __restrict__ tbias,
                      const float* __restrict__ tl1, const float* __restrict__ tl2,
                      float* __restrict__ wgt) {
    __shared__ float h2[WAYN][SEMN / 2];
    __shared__ float wv[WAYN][TN];
    for (int idx = threadIdx.x; idx < WAYN * (SEMN / 2); idx += blockDim.x) {
        int w = idx / (SEMN / 2), i = idx % (SEMN / 2);
        float acc = bt1[i];
        const float* lv = l2v + (size_t)w * SEMN;
        const float* wr = wt1 + (size_t)i * SEMN;
        for (int s = 0; s < SEMN; ++s) acc += lv[s] * wr[s];
        h2[w][i] = lreluf(acc);
    }
    __syncthreads();
    float a1 = tl1[0] * tl1[0], a2 = tl2[0] * tl2[0];
    for (int idx = threadIdx.x; idx < WAYN * TN; idx += blockDim.x) {
        int w = idx / TN, t = idx % TN;
        float acc = bt2[t];
        for (int i = 0; i < SEMN / 2; ++i) acc += h2[w][i] * wt2[t * (SEMN / 2) + i];
        wv[w][t] = a1 * sigmoidf(tbias[t]) + a2 * sigmoidf(acc);
    }
    __syncthreads();
    if (threadIdx.x < WAYN) {
        int w = threadIdx.x;
        float s = 0;
        for (int t = 0; t < TN; ++t) s += wv[w][t];
        float inv = 1.0f / s;
        for (int t = 0; t < TN; ++t) wgt[w * TN + t] = wv[w][t] * inv;
    }
}

// per-pixel inverse channel-norms for support images
__global__ void k_sptnorm(const float* __restrict__ spt, float* __restrict__ invn) {
    int img = blockIdx.x, p = threadIdx.x;
    if (p < NPIX) {
        const float* base = spt + (size_t)img * DIMN * NPIX + p;
        float ss = 0;
        for (int c = 0; c < DIMN; ++c) { float v = base[(size_t)c * NPIX]; ss += v * v; }
        invn[img * NPIX + p] = 1.0f / fmaxf(sqrtf(ss), 1e-12f);
    }
}

// support attention pooling -> spt_tn[img][t][c]   (spatial act-norm cancels under L2n)
__global__ void k_sptpool(const float* __restrict__ spt, const float* __restrict__ invn,
                          const float* __restrict__ filtn, float* __restrict__ spt_tn) {
    int img = blockIdx.x / TN, t = blockIdx.x % TN;
    int w = img % WAYN;
    __shared__ float act[NPIX];
    __shared__ float pooled[DIMN];
    __shared__ float red[16];
    const float* im = spt + (size_t)img * DIMN * NPIX;
    const float* fn = filtn + ((size_t)t * WAYN + w) * DIMN;
    const float* iv = invn + (size_t)img * NPIX;
    for (int p = threadIdx.x; p < NPIX; p += blockDim.x) {
        float acc = 0;
        for (int c = 0; c < DIMN; ++c) acc += im[(size_t)c * NPIX + p] * fn[c];
        act[p] = sigmoidf(acc * iv[p]);
    }
    __syncthreads();
    float ssl = 0;
    for (int c = threadIdx.x; c < DIMN; c += blockDim.x) {
        const float* row = im + (size_t)c * NPIX;
        float acc = 0;
        for (int p = 0; p < NPIX; ++p) acc += row[p] * iv[p] * act[p];
        pooled[c] = acc;
        ssl += acc * acc;
    }
    float ss = block_reduce_sum(ssl, red);
    float inv = 1.0f / fmaxf(sqrtf(ss), 1e-12f);
    for (int c = threadIdx.x; c < DIMN; c += blockDim.x)
        spt_tn[((size_t)img * TN + t) * DIMN + c] = pooled[c] * inv;
}

// qe = l2n(spt_tn @ wq^T + bq)
__global__ void k_qe(const float* __restrict__ spt_tn, const float* __restrict__ wq,
                     const float* __restrict__ bq, float* __restrict__ qe) {
    int img = blockIdx.x / TN, t = blockIdx.x % TN;
    __shared__ float sv[DIMN];
    __shared__ float ov[DIMN];
    __shared__ float red[16];
    for (int c = threadIdx.x; c < DIMN; c += blockDim.x)
        sv[c] = spt_tn[((size_t)img * TN + t) * DIMN + c];
    __syncthreads();
    float ssl = 0;
    for (int d = threadIdx.x; d < DIMN; d += blockDim.x) {
        float acc = bq[t * DIMN + d];
        const float* wr = wq + ((size_t)t * DIMN + d) * DIMN;
        for (int c = 0; c < DIMN; ++c) acc += sv[c] * wr[c];
        ov[d] = acc;
        ssl += acc * acc;
    }
    float ss = block_reduce_sum(ssl, red);
    float inv = 1.0f / fmaxf(sqrtf(ss), 1e-12f);
    for (int d = threadIdx.x; d < DIMN; d += blockDim.x)
        qe[((size_t)img * TN + t) * DIMN + d] = ov[d] * inv;
}

// normalize wkeys rows
__global__ void k_wkn(const float* __restrict__ wkeys, float* __restrict__ wkn) {
    size_t idx = blockIdx.x;  // t*NCLS + k
    const float* row = wkeys + idx * DIMN;
    __shared__ float red[16];
    float ssl = 0;
    for (int d = threadIdx.x; d < DIMN; d += blockDim.x) { float v = row[d]; ssl += v * v; }
    float ss = block_reduce_sum(ssl, red);
    float inv = 1.0f / fmaxf(sqrtf(ss), 1e-12f);
    for (int d = threadIdx.x; d < DIMN; d += blockDim.x) wkn[idx * DIMN + d] = row[d] * inv;
}

// normalize weight_base rows, write transposed wbT[t][d][k]
__global__ void k_wbt(const float* __restrict__ wb, float* __restrict__ wbT) {
    int k = blockIdx.x / TN, t = blockIdx.x % TN;
    const float* row = wb + ((size_t)k * TN + t) * DIMN;
    __shared__ float red[16];
    float ssl = 0;
    for (int d = threadIdx.x; d < DIMN; d += blockDim.x) { float v = row[d]; ssl += v * v; }
    float ss = block_reduce_sum(ssl, red);
    float inv = 1.0f / fmaxf(sqrtf(ss), 1e-12f);
    for (int d = threadIdx.x; d < DIMN; d += blockDim.x)
        wbT[((size_t)t * DIMN + d) * NCLSN + k] = row[d] * inv;
}

// att = softmax(scale * qe @ wk_n^T); wtb = att @ wb_n
__global__ void k_att(const float* __restrict__ qe, const float* __restrict__ wkn,
                      const float* __restrict__ wbT, const float* __restrict__ scale_att,
                      float* __restrict__ wtb) {
    int img = blockIdx.x / TN, t = blockIdx.x % TN;
    __shared__ float q[DIMN];
    __shared__ float att[NCLSN];
    __shared__ float red[16];
    for (int d = threadIdx.x; d < DIMN; d += blockDim.x)
        q[d] = qe[((size_t)img * TN + t) * DIMN + d];
    __syncthreads();
    float sc = scale_att[t];
    for (int k = threadIdx.x; k < NCLSN; k += blockDim.x) {
        const float* row = wkn + ((size_t)t * NCLSN + k) * DIMN;
        float acc = 0;
        for (int d = 0; d < DIMN; ++d) acc += q[d] * row[d];
        att[k] = sc * acc;
    }
    __syncthreads();
    float ml = -3.0e38f;
    for (int k = threadIdx.x; k < NCLSN; k += blockDim.x) ml = fmaxf(ml, att[k]);
    float m = block_reduce_max(ml, red);
    float sl = 0;
    for (int k = threadIdx.x; k < NCLSN; k += blockDim.x) {
        float e = expf(att[k] - m);
        att[k] = e;
        sl += e;
    }
    float s = block_reduce_sum(sl, red);
    float invs = 1.0f / s;
    for (int k = threadIdx.x; k < NCLSN; k += blockDim.x) att[k] *= invs;
    __syncthreads();
    for (int d = threadIdx.x; d < DIMN; d += blockDim.x) {
        const float* col = wbT + ((size_t)t * DIMN + d) * NCLSN;
        float acc = 0;
        for (int k = 0; k < NCLSN; ++k) acc += att[k] * col[k];
        wtb[((size_t)img * TN + t) * DIMN + d] = acc;
    }
}

// proto = l2n(mean_s(lamn^2*spt_tn + lamb^2*wtb)) -> Wglob rows 160+wt (bf16)
__global__ void k_proto(const float* __restrict__ spt_tn, const float* __restrict__ wtb,
                        const float* __restrict__ lamn, const float* __restrict__ lamb,
                        __bf16* __restrict__ wglob) {
    int wt = blockIdx.x, w = wt / TN, t = wt % TN;
    float ln = lamn[t] * lamn[t], lb = lamb[t] * lamb[t];
    __shared__ float comb[DIMN];
    __shared__ float red[16];
    float ssl = 0;
    for (int d = threadIdx.x; d < DIMN; d += blockDim.x) {
        float acc = 0;
        for (int s = 0; s < SHOTN; ++s) {
            size_t idx = (((size_t)(s * WAYN + w)) * TN + t) * DIMN + d;
            acc += ln * spt_tn[idx] + lb * wtb[idx];
        }
        acc *= (1.0f / SHOTN);
        comb[d] = acc;
        ssl += acc * acc;
    }
    float ss = block_reduce_sum(ssl, red);
    float inv = 1.0f / fmaxf(sqrtf(ss), 1e-12f);
    for (int d = threadIdx.x; d < DIMN; d += blockDim.x)
        wglob[((size_t)(NWT + wt)) * DIMN + d] = (__bf16)(comb[d] * inv);
}

// ---------------- main query kernel: 4 WMMA GEMM passes per query ----------------
__global__ void __launch_bounds__(256, 1)
k_query(const float* __restrict__ qry, const __bf16* __restrict__ wglob,
        const float* __restrict__ wgt, float* __restrict__ out) {
    extern __shared__ char smem[];
    __bf16* Abf = (__bf16*)smem;                          // [PPAD][A_RS]
    __bf16* Gt  = (__bf16*)(smem + SM_ABF);               // [PPAD][G_RS]
    __bf16* aT  = (__bf16*)(smem + SM_ABF + SM_GT);       // [NWT][G_RS]
    float*  fsm = (float*)(smem + SM_ABF + SM_GT + SM_AT);
    float* invn  = fsm;             // PPAD
    float* numer = invn + PPAD;     // NWT
    float* dsq   = numer + NWT;     // NWT
    float* simv  = dsq + NWT;       // NWT
    // (+16 floats reduction scratch unused here)

    int q = blockIdx.x;
    int tid = threadIdx.x;
    int lane = tid & 31;
    int wave = tid >> 5;
    const float* im = qry + (size_t)q * DIMN * NPIX;

    // 1) per-pixel inverse channel norms
    if (tid < PPAD) {
        float ss = 0;
        if (tid < NPIX)
            for (int c = 0; c < DIMN; ++c) { float v = im[(size_t)c * NPIX + tid]; ss += v * v; }
        invn[tid] = (tid < NPIX) ? 1.0f / fmaxf(sqrtf(ss), 1e-12f) : 0.0f;
    }
    for (int i = tid; i < NWT; i += 256) { numer[i] = 0.0f; dsq[i] = 0.0f; }
    __syncthreads();

    // 2) build normalized, transposed, zero-padded A (bf16): A[p][c] = qry[c][p]*invn[p]
    for (int idx = tid; idx < PPAD * DIMN; idx += 256) {
        int p = idx & (PPAD - 1);
        int c = idx >> 7;
        float v = (p < NPIX) ? im[(size_t)c * NPIX + p] * invn[p] : 0.0f;
        Abf[(size_t)p * A_RS + c] = (__bf16)v;
    }
    __syncthreads();

    // Pass 1: D = A x filt_n^T  (cols 0..159)  -> a = sigmoid(D) into aT[wt][p]
    for (int tile = wave; tile < 80; tile += 8) {
        int row0 = (tile & 7) * 16, col0 = (tile >> 3) * 16;
        v8f acc = {};
        for (int kk = 0; kk < DIMN; kk += 32) {
            v16bf af = load_frag_a(Abf, row0, kk, A_RS, lane);
            v16bf bf = load_frag_b(wglob, col0, kk, DIMN, lane);
            acc = __builtin_amdgcn_wmma_f32_16x16x32_bf16(false, af, false, bf,
                                                          (short)0, acc, false, false);
        }
        int col = col0 + (lane & 15);
        int rbase = row0 + ((lane >> 4) << 3);
#pragma unroll
        for (int v = 0; v < 8; ++v) {
            int p = rbase + v;
            float a = (p < NPIX) ? sigmoidf(acc[v]) : 0.0f;
            aT[(size_t)col * G_RS + p] = (__bf16)a;
        }
    }
    __syncthreads();

    // Pass 2: P = A x proto^T -> numer[wt] += sum_p a[p,wt]*P[p,wt]
    for (int tile = wave; tile < 80; tile += 8) {
        int row0 = (tile & 7) * 16, col0 = (tile >> 3) * 16;
        v8f acc = {};
        for (int kk = 0; kk < DIMN; kk += 32) {
            v16bf af = load_frag_a(Abf, row0, kk, A_RS, lane);
            v16bf bf = load_frag_b(wglob + (size_t)NWT * DIMN, col0, kk, DIMN, lane);
            acc = __builtin_amdgcn_wmma_f32_16x16x32_bf16(false, af, false, bf,
                                                          (short)0, acc, false, false);
        }
        int col = col0 + (lane & 15);
        int rbase = row0 + ((lane >> 4) << 3);
        float part = 0;
#pragma unroll
        for (int v = 0; v < 8; ++v)
            part += acc[v] * (float)aT[(size_t)col * G_RS + rbase + v];
        atomicAdd(&numer[col], part);
    }
    __syncthreads();

    // Pass 3: Gram G = A x A^T (symmetric) -> Gt (bf16)
    for (int tile = wave; tile < 64; tile += 8) {
        int row0 = (tile & 7) * 16, col0 = (tile >> 3) * 16;
        v8f acc = {};
        for (int kk = 0; kk < DIMN; kk += 32) {
            v16bf af = load_frag_a(Abf, row0, kk, A_RS, lane);
            v16bf bf = load_frag_b(Abf, col0, kk, A_RS, lane);
            acc = __builtin_amdgcn_wmma_f32_16x16x32_bf16(false, af, false, bf,
                                                          (short)0, acc, false, false);
        }
        int col = col0 + (lane & 15);
        int rbase = row0 + ((lane >> 4) << 3);
#pragma unroll
        for (int v = 0; v < 8; ++v)
            Gt[(size_t)col * G_RS + rbase + v] = (__bf16)acc[v];
    }
    __syncthreads();

    // Pass 4: Tm = G x a -> dsq[wt] += sum_p a[p,wt]*Tm[p,wt]   (= ||pooled||^2)
    for (int tile = wave; tile < 80; tile += 8) {
        int row0 = (tile & 7) * 16, col0 = (tile >> 3) * 16;
        v8f acc = {};
        for (int kk = 0; kk < PPAD; kk += 32) {
            v16bf af = load_frag_a(Gt, row0, kk, G_RS, lane);  // symmetric: Gt rows == G rows
            v16bf bf = load_frag_b(aT, col0, kk, G_RS, lane);
            acc = __builtin_amdgcn_wmma_f32_16x16x32_bf16(false, af, false, bf,
                                                          (short)0, acc, false, false);
        }
        int col = col0 + (lane & 15);
        int rbase = row0 + ((lane >> 4) << 3);
        float part = 0;
#pragma unroll
        for (int v = 0; v < 8; ++v)
            part += acc[v] * (float)aT[(size_t)col * G_RS + rbase + v];
        atomicAdd(&dsq[col], part);
    }
    __syncthreads();

    // sim = numer / max(||pooled||, eps);   logits[w] = sum_t sim*wgt / TEMP
    for (int i = tid; i < NWT; i += 256) {
        float den = fmaxf(sqrtf(fmaxf(dsq[i], 0.0f)), 1e-12f);
        simv[i] = numer[i] / den;
    }
    __syncthreads();
    if (tid < WAYN) {
        float acc = 0;
        for (int t = 0; t < TN; ++t) acc += simv[tid * TN + t] * wgt[tid * TN + t];
        out[(size_t)q * WAYN + tid] = acc * 2.0f;  // / TEMP(0.5)
    }
}

// ---------------- host launcher ----------------
extern "C" void kernel_launch(void* const* d_in, const int* in_sizes, int n_in,
                              void* d_out, int out_size, void* d_ws, size_t ws_size,
                              hipStream_t stream) {
    (void)in_sizes; (void)n_in; (void)out_size; (void)ws_size;
    const float* spt    = (const float*)d_in[0];
    const float* qry    = (const float*)d_in[1];
    const float* l2v    = (const float*)d_in[2];
    const float* w1     = (const float*)d_in[3];
    const float* b1     = (const float*)d_in[4];
    const float* w2     = (const float*)d_in[5];
    const float* b2     = (const float*)d_in[6];
    const float* wbase  = (const float*)d_in[7];
    const float* wq     = (const float*)d_in[8];
    const float* bq     = (const float*)d_in[9];
    const float* wkeys  = (const float*)d_in[10];
    const float* scatt  = (const float*)d_in[11];
    const float* lamn   = (const float*)d_in[12];
    const float* lamb   = (const float*)d_in[13];
    const float* wt1    = (const float*)d_in[14];
    const float* bt1    = (const float*)d_in[15];
    const float* wt2    = (const float*)d_in[16];
    const float* bt2    = (const float*)d_in[17];
    const float* tbias  = (const float*)d_in[18];
    const float* tl1    = (const float*)d_in[19];
    const float* tl2    = (const float*)d_in[20];

    char* ws = (char*)d_ws;
    size_t off = 0;
    float*  filtn  = (float*)(ws + off);  off += (size_t)TN * WAYN * DIMN * 4;        // 409600
    __bf16* wglob  = (__bf16*)(ws + off); off += (size_t)2 * NWT * DIMN * 2;          // 409600
    float*  sptinv = (float*)(ws + off);  off += (size_t)SHOTN * WAYN * NPIX * 4;     // 48400
    float*  spttn  = (float*)(ws + off);  off += (size_t)SHOTN * WAYN * TN * DIMN * 4;
    float*  qe     = (float*)(ws + off);  off += (size_t)SHOTN * WAYN * TN * DIMN * 4;
    float*  wkn    = (float*)(ws + off);  off += (size_t)TN * NCLSN * DIMN * 4;
    float*  wbT    = (float*)(ws + off);  off += (size_t)TN * DIMN * NCLSN * 4;
    float*  wtb    = (float*)(ws + off);  off += (size_t)SHOTN * WAYN * TN * DIMN * 4;
    float*  wgt    = (float*)(ws + off);  off += (size_t)WAYN * TN * 4;

    k_filt<<<WAYN * TN, 256, 0, stream>>>(l2v, w1, b1, w2, b2, filtn, wglob);
    k_wgt<<<1, 256, 0, stream>>>(l2v, wt1, bt1, wt2, bt2, tbias, tl1, tl2, wgt);
    k_sptnorm<<<SHOTN * WAYN, 128, 0, stream>>>(spt, sptinv);
    k_sptpool<<<SHOTN * WAYN * TN, 256, 0, stream>>>(spt, sptinv, filtn, spttn);
    k_qe<<<SHOTN * WAYN * TN, 256, 0, stream>>>(spttn, wq, bq, qe);
    k_wkn<<<TN * NCLSN, 256, 0, stream>>>(wkeys, wkn);
    k_wbt<<<NCLSN * TN, 256, 0, stream>>>(wbase, wbT);
    k_att<<<SHOTN * WAYN * TN, 256, 0, stream>>>(qe, wkn, wbT, scatt, wtb);
    k_proto<<<NWT, 256, 0, stream>>>(spttn, wtb, lamn, lamb, wglob);
    k_query<<<NQRY, 256, SMEM_TOTAL, stream>>>(qry, wglob, wgt, (float*)d_out);
}
